// MeNet_CS_11682311045721
// MI455X (gfx1250) — compile-verified, matrix operand order checked by reading, not run on hardware
//
#include <hip/hip_runtime.h>
#include <hip/hip_bf16.h>
#include <math.h>

// ---------------- problem constants ----------------
#define B_    32
#define CIN_  3
#define HWIN_ 225
#define FM_   128
#define NC_   64
#define HO_   75
#define PIXPP (HO_ * HO_)            // 5625
#define NPIX_ (B_ * PIXPP)           // 180000
#define NELT_ (B_ * FM_ * PIXPP)     // 23040000
#define EPS_  1e-5f

// ---------------- workspace layout (float offsets) ----------------
#define WS_W    0          // [0]=W01 [1]=W11
#define WS_WC   16         // 125 floats (wc: 5 x 25)
#define WS_T1   160        // 800 (pooled x, mean over C)
#define WS_M1   960        // 800
#define WS_T2   1760       // 800 (pooled xc, mean over C)
#define WS_M2   2560       // 800
#define WS_EA1  3360       // 128  c1 BN scale
#define WS_EB1  3488       // 128  c1 BN shift (bias folded)
#define WS_EA2  3616       // 128  c2 BN scale
#define WS_EB2  3744       // 128  c2 BN shift (bias folded)
#define WS_SC1  3872       // 64 stream1 BN scale
#define WS_SH1  3936       // 64 stream1 BN shift
#define WS_SC2  4000       // 64 stream2 BN scale
#define WS_SH2  4064       // 64 stream2 BN shift
#define WS_P    4352       // 102400 (pooled features 32x128x5x5)
#define WS_WB1  106752     // 8192 floats  = 16384 bf16  (c1 weights, [ic][oc])
#define WS_WB2  114944     // 73728 floats = 147456 bf16 (c2 weights, [tap][ic][oc])
#define WS_BUFA 188672     // 23040000 floats
#define WS_BUFB (WS_BUFA + NELT_)

// ---------------- vector types ----------------
typedef __attribute__((ext_vector_type(16))) __bf16         v16bf;
typedef __attribute__((ext_vector_type(16))) unsigned short v16us;
typedef __attribute__((ext_vector_type(8)))  unsigned short v8us;
typedef __attribute__((ext_vector_type(8)))  float          v8f;

__device__ __forceinline__ unsigned short f2bf(float x) {
  unsigned int u = __float_as_uint(x);
  unsigned int r = (u + 0x7FFFu + ((u >> 16) & 1u)) >> 16;   // RNE
  return (unsigned short)r;
}

// =============== prep: softmax(archi), wc, BN affine folding ===============
__global__ void prep_kernel(
    const float* __restrict__ archi, const float* __restrict__ fcw,
    const float* __restrict__ c1b, const float* __restrict__ c1g,
    const float* __restrict__ c1be, const float* __restrict__ c1m, const float* __restrict__ c1v,
    const float* __restrict__ c2b, const float* __restrict__ c2g,
    const float* __restrict__ c2be, const float* __restrict__ c2m, const float* __restrict__ c2v,
    const float* __restrict__ s1g, const float* __restrict__ s1be,
    const float* __restrict__ s1m, const float* __restrict__ s1v,
    const float* __restrict__ s2g, const float* __restrict__ s2be,
    const float* __restrict__ s2m, const float* __restrict__ s2v,
    float* __restrict__ wsf)
{
  int tid = threadIdx.x;
  if (tid == 0) {
    float e01 = expf(archi[1]), e11 = expf(archi[3]);
    float inv = 1.f / (e01 + e11);
    wsf[WS_W + 0] = e01 * inv;   // W[0,1]
    wsf[WS_W + 1] = e11 * inv;   // W[1,1]
  }
  if (tid < 125) {
    int n = tid / 25, ij = tid % 25;
    float s = 0.f;
    for (int f = 0; f < FM_; ++f) s += fcw[n * 3200 + f * 25 + ij];
    wsf[WS_WC + tid] = s * (1.f / 128.f);
  }
  if (tid < 128) {
    float a1 = c1g[tid] * rsqrtf(c1v[tid] + EPS_);
    wsf[WS_EA1 + tid] = a1;
    wsf[WS_EB1 + tid] = a1 * (c1b[tid] - c1m[tid]) + c1be[tid];
    float a2 = c2g[tid] * rsqrtf(c2v[tid] + EPS_);
    wsf[WS_EA2 + tid] = a2;
    wsf[WS_EB2 + tid] = a2 * (c2b[tid] - c2m[tid]) + c2be[tid];
  }
  if (tid < 64) {
    float a = s1g[tid] * rsqrtf(s1v[tid] + EPS_);
    wsf[WS_SC1 + tid] = a;
    wsf[WS_SH1 + tid] = s1be[tid] - a * s1m[tid];
    float b = s2g[tid] * rsqrtf(s2v[tid] + EPS_);
    wsf[WS_SC2 + tid] = b;
    wsf[WS_SH2 + tid] = s2be[tid] - b * s2m[tid];
  }
}

// =============== weight repack: fp32 OIHW -> bf16 [tap][ic][oc] ===============
__global__ void wprep_kernel(const float* __restrict__ c1w, const float* __restrict__ c2w,
                             unsigned short* __restrict__ wb1,
                             unsigned short* __restrict__ wb2) {
  int idx = blockIdx.x * 256 + threadIdx.x;
  if (idx < 16384) {                       // c1: [ic][oc] <- [oc][ic]
    int ic = idx >> 7, oc = idx & 127;
    wb1[idx] = f2bf(c1w[oc * 128 + ic]);
  }
  int j = idx - 16384;
  if (j >= 0 && j < 147456) {              // c2: [tap][ic][oc] <- [oc][ic][tap]
    int oc  = j & 127;
    int t2  = j >> 7;
    int ic  = t2 & 127;
    int tap = t2 >> 7;
    wb2[j] = f2bf(c2w[(oc * 128 + ic) * 9 + tap]);
  }
}

// =============== adaptive pool of x (225 -> 5), mean over C=3 ===============
__global__ void pool_x_kernel(const float* __restrict__ x, float* __restrict__ t1) {
  int b = blockIdx.x / 25, w = blockIdx.x % 25;
  int i = w / 5, j = w % 5;
  __shared__ float red[128];
  int tid = threadIdx.x;
  float s = 0.f;
  for (int e = tid; e < 3 * 2025; e += 128) {
    int c = e / 2025, rr = e % 2025;
    int r = rr / 45, col = rr % 45;
    s += x[((b * 3 + c) * 225 + i * 45 + r) * 225 + j * 45 + col];
  }
  red[tid] = s; __syncthreads();
  for (int o = 64; o > 0; o >>= 1) { if (tid < o) red[tid] += red[tid + o]; __syncthreads(); }
  if (tid == 0) t1[b * 25 + w] = red[0] * (1.f / 6075.f);
}

// =============== adaptive pool of xc (75 -> 5), mean over C=128 ===============
__global__ void pool_xc_kernel(const float* __restrict__ xc, float* __restrict__ t2) {
  int b = blockIdx.x / 25, w = blockIdx.x % 25;
  int i = w / 5, j = w % 5;
  __shared__ float red[256];
  int tid = threadIdx.x;
  float s = 0.f;
  for (int e = tid; e < 128 * 225; e += 256) {
    int c = e / 225, rr = e % 225;
    int r = rr / 15, col = rr % 15;
    s += xc[((b * 128 + c) * 75 + i * 15 + r) * 75 + j * 15 + col];
  }
  red[tid] = s; __syncthreads();
  for (int o = 128; o > 0; o >>= 1) { if (tid < o) red[tid] += red[tid + o]; __syncthreads(); }
  if (tid == 0) t2[b * 25 + w] = red[0] * (1.f / 28800.f);
}

// =============== attention map: cam=wc*t, min-max norm, mean over classes ===============
__global__ void atten_kernel(const float* __restrict__ t, const float* __restrict__ wc,
                             float* __restrict__ M) {
  if (threadIdx.x != 0) return;
  int b = blockIdx.x;
  float tv[25], ms[25];
  for (int j = 0; j < 25; ++j) { tv[j] = t[b * 25 + j]; ms[j] = 0.f; }
  for (int n = 0; n < 5; ++n) {
    float cam[25], mn = 3.4e38f, mx = -3.4e38f;
    for (int j = 0; j < 25; ++j) { cam[j] = wc[n * 25 + j] * tv[j]; mn = fminf(mn, cam[j]); }
    for (int j = 0; j < 25; ++j) { cam[j] -= mn; mx = fmaxf(mx, cam[j]); }
    float inv = (mx != 0.f) ? 1.f / mx : 0.f;
    for (int j = 0; j < 25; ++j) ms[j] += cam[j] * inv;
  }
  for (int j = 0; j < 25; ++j) M[b * 25 + j] = ms[j] * 0.2f;
}

// =============== stream convs (3 in-ch) + concat + attention scale ===============
__global__ void stream_kernel(const float* __restrict__ x,
                              const float* __restrict__ s1w, const float* __restrict__ s1b,
                              const float* __restrict__ s2w, const float* __restrict__ s2b,
                              const float* __restrict__ wsf, float* __restrict__ xc) {
  int idx = blockIdx.x * 256 + threadIdx.x;
  if (idx >= NELT_) return;
  int b = idx / (FM_ * PIXPP);
  int r = idx - b * FM_ * PIXPP;
  int oc = r / PIXPP;
  int pix = r - oc * PIXPP;
  int oh = pix / HO_, ow = pix - oh * HO_;
  const float* xb = x + (long)b * 3 * 225 * 225;
  float out;
  if (oc < 64) {   // 3x3 stride 3, pad 1
    float acc = 0.f;
    for (int c = 0; c < 3; ++c)
      for (int kh = 0; kh < 3; ++kh) {
        int ih = oh * 3 + kh - 1;
        if (ih < 0 || ih >= 225) continue;
        for (int kw = 0; kw < 3; ++kw) {
          int iw = ow * 3 + kw - 1;
          if (iw < 0 || iw >= 225) continue;
          acc += xb[(c * 225 + ih) * 225 + iw] * s1w[((oc * 3 + c) * 3 + kh) * 3 + kw];
        }
      }
    float v = fmaxf(acc + s1b[oc], 0.f);                    // ReLU then BN
    out = wsf[WS_SC1 + oc] * v + wsf[WS_SH1 + oc];
  } else {         // 3x3 stride 3, pad 3, dil 3
    int o2 = oc - 64;
    float acc = 0.f;
    for (int c = 0; c < 3; ++c)
      for (int kh = 0; kh < 3; ++kh) {
        int ih = oh * 3 + (kh - 1) * 3;
        if (ih < 0 || ih >= 225) continue;
        for (int kw = 0; kw < 3; ++kw) {
          int iw = ow * 3 + (kw - 1) * 3;
          if (iw < 0 || iw >= 225) continue;
          acc += xb[(c * 225 + ih) * 225 + iw] * s2w[((o2 * 3 + c) * 3 + kh) * 3 + kw];
        }
      }
    float v = fmaxf(acc + s2b[o2], 0.f);
    out = wsf[WS_SC2 + o2] * v + wsf[WS_SH2 + o2];
  }
  float fac = 1.f + wsf[WS_W] * wsf[WS_M1 + b * 25 + (oh / 15) * 5 + (ow / 15)];
  xc[idx] = out * fac;
}

// =============== WMMA implicit-GEMM conv, double-buffered pipeline ===============
// out = relu(effA*conv + effB) [* (1 + wsc*att_up)] [+ addsrc]
template <int KS>
__global__ __launch_bounds__(256) void conv_wmma(
    const float* __restrict__ in,   const unsigned short* __restrict__ wb,
    const float* __restrict__ effA, const float* __restrict__ effB,
    const float* __restrict__ addsrc, const float* __restrict__ att,
    const float* __restrict__ wsc,  float* __restrict__ outp)
{
  constexpr int STR  = 40;                // LDS row stride in halves (80B)
  constexpr int TILE = 128 * STR;
  constexpr int NS   = 4 * KS * KS;       // K-steps: 4 ic-chunks per tap
  __shared__ alignas(16) unsigned short As[2 * TILE];
  __shared__ alignas(16) unsigned short Bs[2 * TILE];

  const int tid   = threadIdx.x;
  const int wv    = tid >> 5;
  const int lane  = tid & 31;
  const int hi    = (lane >> 4) & 1;
  const int lidx  = lane & 15;
  const int mbase = blockIdx.x * 128;

  // load-thread mapping (constant across K loop): row = tid&127, k-parity = tid>>7
  const int  mA   = tid & 127;
  const int  half = tid >> 7;             // 0 or 1
  const int  pA   = mbase + mA;
  const bool pAok = (pA < NPIX_);
  int bA = 0, ohA = 0, owA = 0;
  if (pAok) {
    bA = pA / PIXPP;
    int pr = pA - bA * PIXPP;
    ohA = pr / HO_; owA = pr - ohA * HO_;
  }
  const float* inBase = in + (long)bA * FM_ * PIXPP;
  const int ocB = mA;                     // weight oc this thread stages

  float          aReg[16];
  unsigned short bReg[16];

  auto gload = [&](int s) {
    const int tap = s >> 2, kc = s & 3;
    const int dh = tap / KS - KS / 2, dw = tap % KS - KS / 2;
    const int ih = ohA + dh, iw = owA + dw;
    const bool inb = pAok && (unsigned)ih < (unsigned)HO_ && (unsigned)iw < (unsigned)HO_;
    const int soff = ih * HO_ + iw;
    const unsigned short* wrow = wb + (tap * 128 + kc * 32) * 128 + ocB;
    #pragma unroll
    for (int it = 0; it < 16; ++it) {
      int k = half + it * 2;
      aReg[it] = inb ? inBase[(kc * 32 + k) * PIXPP + soff] : 0.f;
      bReg[it] = wrow[k * 128];
    }
  };

  v8f acc[8] = {};
  gload(0);

  for (int s = 0; s < NS; ++s) {
    unsigned short* Ab = As + (s & 1) * TILE;
    unsigned short* Bb = Bs + (s & 1) * TILE;
    // stage current chunk to LDS (bf16)
    #pragma unroll
    for (int it = 0; it < 16; ++it) {
      int k = half + it * 2;
      Ab[mA * STR + k]  = f2bf(aReg[it]);
      Bb[ocB * STR + k] = bReg[it];
    }
    __syncthreads();                       // single barrier per K-step (double buffer)
    if (s + 1 < NS) gload(s + 1);          // prefetch next chunk; overlaps WMMAs below

    // A fragment: lane holds row (wv*16+lidx); K per ISA 16-bit A layout
    const unsigned short* ap = Ab + (wv * 16 + lidx) * STR + hi * 8;
    v8us a0 = *(const v8us*)(ap);
    v8us a1 = *(const v8us*)(ap + 16);
    v16us aus;
    #pragma unroll
    for (int e = 0; e < 8; ++e) { aus[e] = a0[e]; aus[e + 8] = a1[e]; }
    v16bf av = __builtin_bit_cast(v16bf, aus);

    #pragma unroll
    for (int f = 0; f < 8; ++f) {
      const unsigned short* bp = Bb + (f * 16 + lidx) * STR + hi * 8;
      v8us b0 = *(const v8us*)(bp);
      v8us b1 = *(const v8us*)(bp + 16);
      v16us bus;
      #pragma unroll
      for (int e = 0; e < 8; ++e) { bus[e] = b0[e]; bus[e + 8] = b1[e]; }
      v16bf bv = __builtin_bit_cast(v16bf, bus);
      acc[f] = __builtin_amdgcn_wmma_f32_16x16x32_bf16(
          false, av, false, bv, (short)0, acc[f], false, false);
    }
  }

  // ---- epilogue: BN + ReLU (+attention) (+residual add) ----
  const float w1 = att ? *wsc : 0.f;
  #pragma unroll
  for (int f = 0; f < 8; ++f) {
    int n  = f * 16 + lidx;               // output channel
    float sA = effA[n], sB = effB[n];
    #pragma unroll
    for (int r8 = 0; r8 < 8; ++r8) {
      int m = wv * 16 + r8 + hi * 8;
      int p = mbase + m;
      if (p < NPIX_) {
        int b  = p / PIXPP;
        int pr = p - b * PIXPP;
        int oh = pr / HO_, ow = pr - oh * HO_;
        float v = sA * acc[f][r8] + sB;
        v = fmaxf(v, 0.f);
        if (att) v *= 1.f + w1 * att[b * 25 + (oh / 15) * 5 + (ow / 15)];
        int idx = ((b * FM_ + n) * HO_ + oh) * HO_ + ow;
        if (addsrc) v += addsrc[idx];
        outp[idx] = v;
      }
    }
  }
}

// =============== maxpool 2x2 (75->37) + adaptive avg pool (37->5) ===============
__global__ void final_pool_kernel(const float* __restrict__ o, float* __restrict__ p) {
  int bc = blockIdx.x;                    // b*128 + c
  int t = threadIdx.x;
  if (t >= 25) return;
  int i = t / 5, j = t % 5;
  int h0 = (i * 37) / 5, h1 = ((i + 1) * 37 + 4) / 5;
  int w0 = (j * 37) / 5, w1 = ((j + 1) * 37 + 4) / 5;
  const float* base = o + (long)bc * PIXPP;
  float s = 0.f;
  for (int mh = h0; mh < h1; ++mh)
    for (int mw = w0; mw < w1; ++mw) {
      const float* r0 = base + (2 * mh) * HO_ + 2 * mw;
      float m = fmaxf(fmaxf(r0[0], r0[1]), fmaxf(r0[HO_], r0[HO_ + 1]));
      s += m;
    }
  p[bc * 25 + t] = s / (float)((h1 - h0) * (w1 - w0));
}

// =============== final FC: (32,3200) @ (5,3200)^T + bias ===============
__global__ void fc_kernel(const float* __restrict__ p, const float* __restrict__ fcw,
                          const float* __restrict__ fcb, float* __restrict__ out) {
  int b = blockIdx.x / 5, n = blockIdx.x % 5;
  __shared__ float red[256];
  int tid = threadIdx.x;
  float s = 0.f;
  for (int k = tid; k < 3200; k += 256) s += p[b * 3200 + k] * fcw[n * 3200 + k];
  red[tid] = s; __syncthreads();
  for (int o = 128; o > 0; o >>= 1) { if (tid < o) red[tid] += red[tid + o]; __syncthreads(); }
  if (tid == 0) out[b * 5 + n] = red[0] + fcb[n];
}

// =====================================================================
extern "C" void kernel_launch(void* const* d_in, const int* in_sizes, int n_in,
                              void* d_out, int out_size, void* d_ws, size_t ws_size,
                              hipStream_t stream) {
  (void)in_sizes; (void)n_in; (void)out_size; (void)ws_size;
  const float* x    = (const float*)d_in[0];
  const float* arch = (const float*)d_in[1];
  const float* s1w  = (const float*)d_in[2];
  const float* s1b  = (const float*)d_in[3];
  const float* s2w  = (const float*)d_in[4];
  const float* s2b  = (const float*)d_in[5];
  const float* c1w  = (const float*)d_in[6];
  const float* c1b  = (const float*)d_in[7];
  const float* c2w  = (const float*)d_in[8];
  const float* c2b  = (const float*)d_in[9];
  const float* s1g  = (const float*)d_in[10];
  const float* s1be = (const float*)d_in[11];
  const float* s1m  = (const float*)d_in[12];
  const float* s1v  = (const float*)d_in[13];
  const float* s2g  = (const float*)d_in[14];
  const float* s2be = (const float*)d_in[15];
  const float* s2m  = (const float*)d_in[16];
  const float* s2v  = (const float*)d_in[17];
  const float* c1g  = (const float*)d_in[18];
  const float* c1be = (const float*)d_in[19];
  const float* c1m  = (const float*)d_in[20];
  const float* c1v  = (const float*)d_in[21];
  const float* c2g  = (const float*)d_in[22];
  const float* c2be = (const float*)d_in[23];
  const float* c2m  = (const float*)d_in[24];
  const float* c2v  = (const float*)d_in[25];
  const float* fcw  = (const float*)d_in[26];
  const float* fcb  = (const float*)d_in[27];
  float* wsf = (float*)d_ws;
  float* out = (float*)d_out;
  unsigned short* wb1 = (unsigned short*)(wsf + WS_WB1);
  unsigned short* wb2 = (unsigned short*)(wsf + WS_WB2);

  prep_kernel<<<1, 256, 0, stream>>>(arch, fcw,
      c1b, c1g, c1be, c1m, c1v, c2b, c2g, c2be, c2m, c2v,
      s1g, s1be, s1m, s1v, s2g, s2be, s2m, s2v, wsf);
  wprep_kernel<<<(16384 + 147456) / 256, 256, 0, stream>>>(c1w, c2w, wb1, wb2);

  pool_x_kernel<<<B_ * 25, 128, 0, stream>>>(x, wsf + WS_T1);
  atten_kernel<<<B_, 32, 0, stream>>>(wsf + WS_T1, wsf + WS_WC, wsf + WS_M1);

  stream_kernel<<<NELT_ / 256, 256, 0, stream>>>(x, s1w, s1b, s2w, s2b, wsf,
                                                 wsf + WS_BUFA);        // xc -> A

  pool_xc_kernel<<<B_ * 25, 256, 0, stream>>>(wsf + WS_BUFA, wsf + WS_T2);
  atten_kernel<<<B_, 32, 0, stream>>>(wsf + WS_T2, wsf + WS_WC, wsf + WS_M2);

  const int convGrid = (NPIX_ + 127) / 128;   // 1407 tiles of 128 pixels
  // s0 = xc + relu(bn(conv1x1(xc))) * (1 + W11*M2up)           -> B
  conv_wmma<1><<<convGrid, 256, 0, stream>>>(wsf + WS_BUFA, wb1,
      wsf + WS_EA1, wsf + WS_EB1, wsf + WS_BUFA, wsf + WS_M2, wsf + WS_W + 1,
      wsf + WS_BUFB);
  // s1 = s0 + relu(bn(conv3x3(s0)))                            -> A
  conv_wmma<3><<<convGrid, 256, 0, stream>>>(wsf + WS_BUFB, wb2,
      wsf + WS_EA2, wsf + WS_EB2, wsf + WS_BUFB, nullptr, nullptr, wsf + WS_BUFA);
  // s2 = s1 + relu(bn(conv3x3(s1)))                            -> B
  conv_wmma<3><<<convGrid, 256, 0, stream>>>(wsf + WS_BUFA, wb2,
      wsf + WS_EA2, wsf + WS_EB2, wsf + WS_BUFA, nullptr, nullptr, wsf + WS_BUFB);
  // out = relu(bn(conv3x3(s2)))                                -> A
  conv_wmma<3><<<convGrid, 256, 0, stream>>>(wsf + WS_BUFB, wb2,
      wsf + WS_EA2, wsf + WS_EB2, nullptr, nullptr, nullptr, wsf + WS_BUFA);

  final_pool_kernel<<<B_ * FM_, 32, 0, stream>>>(wsf + WS_BUFA, wsf + WS_P);
  fc_kernel<<<B_ * 5, 256, 0, stream>>>(wsf + WS_P, fcw, fcb, out);
}
